// TreeAwareAttention_89326729822324
// MI455X (gfx1250) — compile-verified, matrix operand order checked by reading, not run on hardware
//
#include <hip/hip_runtime.h>
#include <math.h>

// ---------------------------------------------------------------------------
// TreeAwareAttention for MI455X (gfx1250, wave32, WMMA).
// All GEMMs on v_wmma_f32_16x16x32_bf16 (fp32 accumulate), zero LDS:
//  k0: fp32 -> bf16 pre-convert (x, Wq, Wk, Wv, Wo)           ~13 MB, ~0.6 us
//  k1: QKV projection, Q/K head-major [b][h][n][d], V stored TRANSPOSED
//      [b][h][d][n] so attention reads V^T fragments contiguously.
//  k2: flash attention, transposed scores S^T = K·Q^T (softmax state per-lane,
//      P^T pack is lane-local, one shfl_xor(16) per reduction), exp2 domain.
//  k3: output projection ctx @ Wo^T + bo -> fp32.
// Workspace: ~22 MB of bf16 intermediates.
// ---------------------------------------------------------------------------

typedef __attribute__((ext_vector_type(16))) __bf16 v16bf;
typedef __attribute__((ext_vector_type(8)))  __bf16 v8bf;
typedef __attribute__((ext_vector_type(8)))  float  v8f;
typedef __attribute__((ext_vector_type(4)))  float  v4f;
typedef __attribute__((ext_vector_type(4)))  unsigned v4u;

constexpr int B_  = 2;
constexpr int H_  = 8;
constexpr int N_  = 2048;
constexpr int DK_ = 64;
constexpr int DM_ = 512;

__device__ __forceinline__ unsigned short f2bf(float f) {
  union { float f; unsigned u; } c; c.f = f;
  unsigned r = c.u + 0x7FFFu + ((c.u >> 16) & 1u);   // round-to-nearest-even
  return (unsigned short)(r >> 16);
}

__device__ __forceinline__ float fast_exp2(float x) {
#if __has_builtin(__builtin_amdgcn_exp2f)
  return __builtin_amdgcn_exp2f(x);                  // v_exp_f32 (base-2)
#else
  return exp2f(x);
#endif
}

// Build a 16x32 A-operand / 32x16 B-operand bf16 fragment for one lane.
// p = this lane's row (M for A, N for B) of a K=32 contiguous bf16 chunk.
// ISA 7.12.2: lo lanes hold K={0..7,16..23}, hi lanes K={8..15,24..31}.
__device__ __forceinline__ v16bf frag16(const unsigned short* p, int hiH) {
  const v8bf a = *(const v8bf*)(p + (hiH ? 8 : 0));
  const v8bf b = *(const v8bf*)(p + (hiH ? 8 : 0) + 16);
  return __builtin_shufflevector(a, b, 0,1,2,3,4,5,6,7,8,9,10,11,12,13,14,15);
}

// ---------------------------------------------------------------------------
// Kernel 0: fp32 -> bf16 bulk convert (n multiple of 2048). 8 elems/thread.
// ---------------------------------------------------------------------------
__global__ __launch_bounds__(256) void k_tobf16(const float* __restrict__ src,
                                                unsigned short* __restrict__ dst,
                                                int n) {
  const int i = (blockIdx.x * 256 + threadIdx.x) * 8;
  if (i >= n) return;
  const v4f a = *(const v4f*)(src + i);
  const v4f b = *(const v4f*)(src + i + 4);
  union { v4u u; unsigned short s[8]; } pk;
  #pragma unroll
  for (int j = 0; j < 4; ++j) { pk.s[j] = f2bf(a[j]); pk.s[4 + j] = f2bf(b[j]); }
  *(v4u*)(dst + i) = pk.u;
}

// ---------------------------------------------------------------------------
// Kernel 1: fused QKV projection, no LDS. out = x @ W^T + b.
// grid = (B*N/128, H, 3); block = 256 (8 waves x 16 tokens, 64 feats each).
// z==0 -> Q [b][h][n][d], z==1 -> K same, z==2 -> V^T [b][h][d][n] (packed).
// ---------------------------------------------------------------------------
__global__ __launch_bounds__(256) void k_qkv(
    const unsigned short* __restrict__ xb,
    const unsigned short* __restrict__ Wqb, const float* __restrict__ bq,
    const unsigned short* __restrict__ Wkb, const float* __restrict__ bk,
    const unsigned short* __restrict__ Wvb, const float* __restrict__ bv,
    unsigned short* __restrict__ Qb, unsigned short* __restrict__ Kb,
    unsigned short* __restrict__ Vt)
{
  const unsigned short* W; const float* bias; unsigned short* dst;
  if      (blockIdx.z == 0) { W = Wqb; bias = bq; dst = Qb; }
  else if (blockIdx.z == 1) { W = Wkb; bias = bk; dst = Kb; }
  else                      { W = Wvb; bias = bv; dst = Vt; }

  const int row0 = blockIdx.x * 128;          // token tile base (0..4095)
  const int h    = blockIdx.y;
  const int f0   = h * DK_;
  const int tid  = threadIdx.x;
  const int wv   = tid >> 5;
  const int lane = tid & 31;
  const int l16  = lane & 15;
  const int hiH  = lane >> 4;

  v8f acc[4];
  #pragma unroll
  for (int t = 0; t < 4; ++t)
    #pragma unroll
    for (int i = 0; i < 8; ++i) acc[t][i] = 0.0f;

  // per-lane fragment row pointers: all contiguous 16B loads
  const unsigned short* arow = xb + (size_t)(row0 + wv * 16 + l16) * DM_;
  const unsigned short* wrow[4];
  #pragma unroll
  for (int t = 0; t < 4; ++t) wrow[t] = W + (size_t)(f0 + t * 16 + l16) * DM_;

  for (int kk = 0; kk < DM_; kk += 32) {
    const v16bf a = frag16(arow + kk, hiH);          // A: 16 tokens x 32 k
    #pragma unroll
    for (int t = 0; t < 4; ++t) {
      const v16bf bf = frag16(wrow[t] + kk, hiH);    // B: 32 k x 16 feats
      acc[t] = __builtin_amdgcn_wmma_f32_16x16x32_bf16(
          false, a, false, bf, (short)0, acc[t], false, false);
    }
  }

  const int bb = row0 >> 11;                 // batch (128-tiles never straddle)
  const int n0 = (row0 & (N_ - 1)) + wv * 16 + hiH * 8;

  if (blockIdx.z == 2) {
    // V^T epilogue: lane owns feature d; its 8 accs are CONSECUTIVE tokens ->
    // one packed 16B store per tile into [b][h][d][n].
    #pragma unroll
    for (int t = 0; t < 4; ++t) {
      const int d = t * 16 + l16;
      const float bvv = bias[f0 + d];
      union { v4u u; unsigned short s[8]; } pk;
      #pragma unroll
      for (int i = 0; i < 8; ++i) pk.s[i] = f2bf(acc[t][i] + bvv);
      *(v4u*)(dst + ((size_t)(bb * H_ + h) * DK_ + d) * N_ + n0) = pk.u;
    }
  } else {
    // Q/K epilogue: head-major [b][h][n][d] (token-strided scalar stores)
    #pragma unroll
    for (int t = 0; t < 4; ++t) {
      const int d = t * 16 + l16;
      const float bvv = bias[f0 + d];
      #pragma unroll
      for (int i = 0; i < 8; ++i) {
        dst[((size_t)(bb * H_ + h) * N_ + n0 + i) * DK_ + d] = f2bf(acc[t][i] + bvv);
      }
    }
  }
}

// ---------------------------------------------------------------------------
// Kernel 2: flash attention, transposed scores, LDS-free, exp2-domain softmax.
// grid = (N/128, H, B); block = 256 (8 waves, 16 q-rows each).
// Per 32-key block: 4 score WMMAs + 4 O WMMAs.
// ---------------------------------------------------------------------------
__global__ __launch_bounds__(256) void k_attn(
    const unsigned short* __restrict__ Q,
    const unsigned short* __restrict__ K,
    const unsigned short* __restrict__ Vt,     // [b][h][d][n]
    const float* __restrict__ Dm, const float* __restrict__ Am,
    const float* __restrict__ wd, const float* __restrict__ wa,
    unsigned short* __restrict__ ctx)
{
  const int qt = blockIdx.x, h = blockIdx.y, b = blockIdx.z;
  const int tid  = threadIdx.x;
  const int wv   = tid >> 5;
  const int lane = tid & 31;
  const int l16  = lane & 15;
  const int hiH  = lane >> 4;
  const int qrow = qt * 128 + wv * 16 + l16;       // this lane's query row

  const unsigned short* Qh  = Q  + (size_t)(b * H_ + h) * N_ * DK_;
  const unsigned short* Kh  = K  + (size_t)(b * H_ + h) * N_ * DK_;
  const unsigned short* Vth = Vt + (size_t)(b * H_ + h) * DK_ * N_;
  const float* Dq = Dm + ((size_t)b * N_ + qrow) * N_;
  const float* Aq = Am + ((size_t)b * N_ + qrow) * N_;

  // exp2 domain: fold log2(e) into scale and bias weights (exact rescale)
  const float LOG2E = 1.4426950408889634f;
  const float sc2 = 0.125f * LOG2E;                // (1/sqrt(64)) * log2(e)
  const float wd2 = wd[h] * LOG2E, wa2 = wa[h] * LOG2E;

  // Q as B-operand (lane = q, packed along d), loaded once
  const v16bf qb0 = frag16(Qh + (size_t)qrow * DK_ +  0, hiH);
  const v16bf qb1 = frag16(Qh + (size_t)qrow * DK_ + 32, hiH);

  // V^T A-operand row pointers (lane = d, contiguous along keys)
  const unsigned short* vrow[4];
  #pragma unroll
  for (int t = 0; t < 4; ++t) vrow[t] = Vth + (size_t)(t * 16 + l16) * N_;

  float m0 = -INFINITY, l0 = 0.0f;                  // log2-domain running state
  v8f o[4];                                         // O^T tiles: [d-tile][q]
  #pragma unroll
  for (int t = 0; t < 4; ++t)
    #pragma unroll
    for (int i = 0; i < 8; ++i) o[t][i] = 0.0f;

  for (int kb = 0; kb < N_; kb += 32) {
    // ---- scores S^T (M=key, N=q): two 16-key tiles, K=64 contraction ----
    v8f s0, s1;
    #pragma unroll
    for (int i = 0; i < 8; ++i) { s0[i] = 0.0f; s1[i] = 0.0f; }
    {
      const size_t k0 = (size_t)(kb + l16) * DK_;
      v16bf ka = frag16(Kh + k0, hiH);
      s0 = __builtin_amdgcn_wmma_f32_16x16x32_bf16(false, ka, false, qb0, (short)0, s0, false, false);
      ka = frag16(Kh + k0 + 32, hiH);
      s0 = __builtin_amdgcn_wmma_f32_16x16x32_bf16(false, ka, false, qb1, (short)0, s0, false, false);
      const size_t k1 = (size_t)(kb + 16 + l16) * DK_;
      ka = frag16(Kh + k1, hiH);
      s1 = __builtin_amdgcn_wmma_f32_16x16x32_bf16(false, ka, false, qb0, (short)0, s1, false, false);
      ka = frag16(Kh + k1 + 32, hiH);
      s1 = __builtin_amdgcn_wmma_f32_16x16x32_bf16(false, ka, false, qb1, (short)0, s1, false, false);
    }

    // ---- fused tree biases: keys contiguous per lane (q fixed per lane) ----
    const float* dp = Dq + kb + hiH * 8;
    const float* ap = Aq + kb + hiH * 8;
    __builtin_prefetch(dp + 32, 0, 1);              // global_prefetch_b8
    __builtin_prefetch(ap + 32, 0, 1);
    const v4f d00 = *(const v4f*)dp,        d01 = *(const v4f*)(dp + 4);
    const v4f d10 = *(const v4f*)(dp + 16), d11 = *(const v4f*)(dp + 20);
    const v4f a00 = *(const v4f*)ap,        a01 = *(const v4f*)(ap + 4);
    const v4f a10 = *(const v4f*)(ap + 16), a11 = *(const v4f*)(ap + 20);
    #pragma unroll
    for (int i = 0; i < 8; ++i) {
      const float dv0 = (i < 4) ? d00[i] : d01[i - 4];
      const float av0 = (i < 4) ? a00[i] : a01[i - 4];
      const float dv1 = (i < 4) ? d10[i] : d11[i - 4];
      const float av1 = (i < 4) ? a10[i] : a11[i - 4];
      s0[i] = s0[i] * sc2 + wd2 * dv0 + wa2 * av0;  // log2-domain logits
      s1[i] = s1[i] * sc2 + wd2 * dv1 + wa2 * av1;
    }

    // ---- online softmax (per-q state in lanes; one shfl_xor(16) each) ----
    float bm = -INFINITY;
    #pragma unroll
    for (int i = 0; i < 8; ++i) bm = fmaxf(bm, fmaxf(s0[i], s1[i]));
    bm = fmaxf(bm, __shfl_xor(bm, 16, 32));
    const float mn   = fmaxf(m0, bm);
    const float corr = fast_exp2(m0 - mn);          // v_exp_f32, no extra mul
    float ps = 0.0f;
    #pragma unroll
    for (int i = 0; i < 8; ++i) {
      s0[i] = fast_exp2(s0[i] - mn); ps += s0[i];
      s1[i] = fast_exp2(s1[i] - mn); ps += s1[i];
    }
    ps += __shfl_xor(ps, 16, 32);
    l0 = l0 * corr + ps;
    m0 = mn;
    #pragma unroll
    for (int t = 0; t < 4; ++t)
      #pragma unroll
      for (int i = 0; i < 8; ++i) o[t][i] *= corr;

    // ---- pack P^T into B-operand (lane-local: C-layout == B-layout) ----
    union { v16bf v; unsigned short s[16]; } pb;
    #pragma unroll
    for (int i = 0; i < 8; ++i) {
      pb.s[i]     = f2bf(s0[i]);   // keys kb+{0..7 | 8..15}
      pb.s[8 + i] = f2bf(s1[i]);   // keys kb+{16..23 | 24..31}
    }

    // ---- O^T += V^T · P^T  (4 d-tiles; V^T rows straight from global) ----
    #pragma unroll
    for (int t = 0; t < 4; ++t) {
      const v16bf va = frag16(vrow[t] + kb, hiH);
      o[t] = __builtin_amdgcn_wmma_f32_16x16x32_bf16(
          false, va, false, pb.v, (short)0, o[t], false, false);
    }
  }

  // ---- normalize and store ctx[b][n][h*64+d] (pack bf16 pairs -> b32) ----
  const float inv = 1.0f / l0;
  unsigned short* crow = ctx + ((size_t)b * N_ + qrow) * DM_ + h * DK_;
  #pragma unroll
  for (int t = 0; t < 4; ++t) {
    #pragma unroll
    for (int i = 0; i < 8; i += 2) {
      const int d = t * 16 + i + hiH * 8;           // even -> 4B aligned
      const unsigned lo = f2bf(o[t][i] * inv);
      const unsigned hh = f2bf(o[t][i + 1] * inv);
      *(unsigned*)(crow + d) = lo | (hh << 16);
    }
  }
}

// ---------------------------------------------------------------------------
// Kernel 3: output projection  out = ctx @ Wo^T + bo  (fp32 out), no LDS.
// grid = (B*N/128, 8); block = 256.
// ---------------------------------------------------------------------------
__global__ __launch_bounds__(256) void k_oproj(
    const unsigned short* __restrict__ ctx,
    const unsigned short* __restrict__ Wob, const float* __restrict__ bo,
    float* __restrict__ out)
{
  const int row0 = blockIdx.x * 128;
  const int f0   = blockIdx.y * 64;
  const int tid  = threadIdx.x;
  const int wv   = tid >> 5;
  const int lane = tid & 31;
  const int l16  = lane & 15;
  const int hiH  = lane >> 4;

  v8f acc[4];
  #pragma unroll
  for (int t = 0; t < 4; ++t)
    #pragma unroll
    for (int i = 0; i < 8; ++i) acc[t][i] = 0.0f;

  const unsigned short* arow = ctx + (size_t)(row0 + wv * 16 + l16) * DM_;
  const unsigned short* wrow[4];
  #pragma unroll
  for (int t = 0; t < 4; ++t) wrow[t] = Wob + (size_t)(f0 + t * 16 + l16) * DM_;

  for (int kk = 0; kk < DM_; kk += 32) {
    const v16bf a = frag16(arow + kk, hiH);
    #pragma unroll
    for (int t = 0; t < 4; ++t) {
      const v16bf bf = frag16(wrow[t] + kk, hiH);
      acc[t] = __builtin_amdgcn_wmma_f32_16x16x32_bf16(
          false, a, false, bf, (short)0, acc[t], false, false);
    }
  }

  #pragma unroll
  for (int t = 0; t < 4; ++t) {
    const int f = f0 + t * 16 + l16;
    const float bvv = bo[f];
    #pragma unroll
    for (int i = 0; i < 8; ++i) {
      const int tok = row0 + wv * 16 + i + hiH * 8;
      out[(size_t)tok * DM_ + f] = acc[t][i] + bvv;
    }
  }
}

// ---------------------------------------------------------------------------
extern "C" void kernel_launch(void* const* d_in, const int* in_sizes, int n_in,
                              void* d_out, int out_size, void* d_ws, size_t ws_size,
                              hipStream_t stream) {
  (void)in_sizes; (void)n_in; (void)out_size; (void)ws_size;
  const float* x  = (const float*)d_in[0];
  const float* Dm = (const float*)d_in[1];
  const float* Am = (const float*)d_in[2];
  const float* Wq = (const float*)d_in[3];
  const float* bq = (const float*)d_in[4];
  const float* Wk = (const float*)d_in[5];
  const float* bk = (const float*)d_in[6];
  const float* Wv = (const float*)d_in[7];
  const float* bv = (const float*)d_in[8];
  const float* Wo = (const float*)d_in[9];
  const float* bo = (const float*)d_in[10];
  const float* wd = (const float*)d_in[11];
  const float* wa = (const float*)d_in[12];

  const size_t NX = (size_t)B_ * N_ * DM_;     // 2,097,152 (x / ctx elems)
  const size_t NW = (size_t)DM_ * DM_;         //   262,144 (weight elems)
  const size_t NH = (size_t)B_ * H_ * N_ * DK_;// 2,097,152 (per head-tensor)

  unsigned short* xbf = (unsigned short*)d_ws;
  unsigned short* Wqb = xbf + NX;
  unsigned short* Wkb = Wqb + NW;
  unsigned short* Wvb = Wkb + NW;
  unsigned short* Wob = Wvb + NW;
  unsigned short* Qb  = Wob + NW;
  unsigned short* Kb  = Qb + NH;
  unsigned short* Vt  = Kb + NH;               // [b][h][d][n]
  unsigned short* ctx = Vt + NH;               // [b][n][512]

  // k0: bulk fp32 -> bf16 (2048 elems per block)
  k_tobf16<<<dim3((unsigned)(NX / 2048)), 256, 0, stream>>>(x,  xbf, (int)NX);
  k_tobf16<<<dim3((unsigned)(NW / 2048)), 256, 0, stream>>>(Wq, Wqb, (int)NW);
  k_tobf16<<<dim3((unsigned)(NW / 2048)), 256, 0, stream>>>(Wk, Wkb, (int)NW);
  k_tobf16<<<dim3((unsigned)(NW / 2048)), 256, 0, stream>>>(Wv, Wvb, (int)NW);
  k_tobf16<<<dim3((unsigned)(NW / 2048)), 256, 0, stream>>>(Wo, Wob, (int)NW);

  k_qkv  <<<dim3((B_ * N_) / 128, H_, 3), 256, 0, stream>>>(
      xbf, Wqb, bq, Wkb, bk, Wvb, bv, Qb, Kb, Vt);
  k_attn <<<dim3(N_ / 128, H_, B_), 256, 0, stream>>>(
      Qb, Kb, Vt, Dm, Am, wd, wa, ctx);
  k_oproj<<<dim3((B_ * N_) / 128, DM_ / 64), 256, 0, stream>>>(
      ctx, Wob, bo, (float*)d_out);
}
